// GraphAttentionLayer_63513976373301
// MI455X (gfx1250) — compile-verified, matrix operand order checked by reading, not run on hardware
//
#include <hip/hip_runtime.h>

typedef __attribute__((ext_vector_type(16))) _Float16    v16h;
typedef __attribute__((ext_vector_type(8)))  float       v8f;
typedef __attribute__((ext_vector_type(4)))  unsigned int v4u;

static constexpr int   BB     = 2;
static constexpr int   SS     = 2048;
static constexpr int   EE     = 512;
static constexpr int   HH     = 8;
static constexpr int   HD     = 64;
static constexpr float SCALE  = 0.125f;   // 1/sqrt(64)
static constexpr float THRESH = 0.1f;

// ---------- fragment helpers (layouts per CDNA5 ISA §7.12.2) ----------
// 16-bit A 16x32: lane&15 = row M, lane>>4 selects K-half; element t holds
// K = (t/8)*16 + (lane>>4)*8 + (t%8)  -> two contiguous 16-byte loads per lane.
// B 32x16 loaded from a row-major [N][K] matrix uses the identical loader
// (lane&15 = column N, per-lane K run), i.e. B[k,n] = W[n][k] = W^T.
__device__ __forceinline__ v16h load_frag(const _Float16* p) {
  union { v16h h; v4u u[2]; } f;
  f.u[0] = *(const v4u*)(p);
  f.u[1] = *(const v4u*)(p + 16);
  return f.h;
}

__device__ __forceinline__ v8f wmma16(v16h a, v16h b, v8f c) {
  return __builtin_amdgcn_wmma_f32_16x16x32_f16(false, a, false, b, (short)0, c,
                                                false, false);
}

// ---------- conversions ----------
__global__ void cvt_f16(const float* __restrict__ in, _Float16* __restrict__ out,
                        int rows, int cols, int ldin) {
  long long i = (long long)blockIdx.x * blockDim.x + threadIdx.x;
  long long total = (long long)rows * cols;
  if (i >= total) return;
  int r = (int)(i / cols), c = (int)(i % cols);
  out[i] = (_Float16)in[(long long)r * ldin + c];
}

// Batched LDS-tiled transpose: in [R][C] -> out [C][R]  (per blockIdx.z batch)
__global__ __launch_bounds__(256) void transpose_f16(
    const _Float16* __restrict__ in, _Float16* __restrict__ out, int R, int C) {
  __shared__ _Float16 tile[32][33];
  long long boff = (long long)blockIdx.z * R * C;
  int c0 = blockIdx.x * 32, r0 = blockIdx.y * 32;
  int tx = threadIdx.x & 31, ty = threadIdx.x >> 5;  // 32 x 8
#pragma unroll
  for (int i = 0; i < 4; ++i) {
    int r = r0 + ty + i * 8;
    tile[ty + i * 8][tx] = in[boff + (long long)r * C + c0 + tx];
  }
  __syncthreads();
#pragma unroll
  for (int i = 0; i < 4; ++i) {
    int c = c0 + ty + i * 8;
    out[boff + (long long)c * R + r0 + tx] = tile[tx][ty + i * 8];
  }
}

// edge_features == 1/S exactly (softmax sums to 1), so fold its Wm1 columns
// into an effective bias.
__global__ void make_bm1eff(const float* __restrict__ Wm1,
                            const float* __restrict__ bm1,
                            float* __restrict__ outb) {
  int e = blockIdx.x * blockDim.x + threadIdx.x;
  if (e >= EE) return;
  float s = 0.f;
#pragma unroll
  for (int h = 0; h < HH; ++h) s += Wm1[(long long)e * (EE + HH) + EE + h];
  outb[e] = bm1[e] + s * (1.0f / (float)SS);
}

// ---------- pass 1: per-(b,h,row) softmax stats (max, 1/sum-exp) ----------
// Two sweeps: (A) pure-max sweep (no exp at all), (B) recompute scores and
// accumulate a single __expf per element. Exp work halved vs online softmax;
// K tiles re-read in sweep B hit L2 (192 MB).
__global__ __launch_bounds__(32) void attn_stats(
    const _Float16* __restrict__ Qh, const _Float16* __restrict__ Kh,
    float2* __restrict__ ml) {
  int it = blockIdx.x, h = blockIdx.y, b = blockIdx.z;
  int lane = threadIdx.x;
  int lr = lane & 15, kh = lane >> 4;
  const _Float16* Qb = Qh + (long long)b * SS * EE;
  const _Float16* Kb = Kh + (long long)b * SS * EE;
  int i0 = it * 16, e0 = h * HD;
  const _Float16* qp = Qb + (long long)(i0 + lr) * EE + e0 + kh * 8;
  v16h qa0 = load_frag(qp);
  v16h qa1 = load_frag(qp + 32);

  float m[8];
#pragma unroll
  for (int r = 0; r < 8; ++r) m[r] = -3.0e38f;
  for (int j0 = 0; j0 < SS; j0 += 16) {     // sweep A: max only
    const _Float16* kp = Kb + (long long)(j0 + lr) * EE + e0 + kh * 8;
    v8f c = {};
    c = wmma16(qa0, load_frag(kp), c);
    c = wmma16(qa1, load_frag(kp + 32), c);
#pragma unroll
    for (int r = 0; r < 8; ++r) m[r] = fmaxf(m[r], c[r] * SCALE);
  }
  // rows 0-7 live in lanes 0-15, rows 8-15 in lanes 16-31
#pragma unroll
  for (int mask = 1; mask <= 8; mask <<= 1) {
#pragma unroll
    for (int r = 0; r < 8; ++r) m[r] = fmaxf(m[r], __shfl_xor(m[r], mask, 32));
  }

  float l[8] = {0.f, 0.f, 0.f, 0.f, 0.f, 0.f, 0.f, 0.f};
  for (int j0 = 0; j0 < SS; j0 += 16) {     // sweep B: sum of exp
    const _Float16* kp = Kb + (long long)(j0 + lr) * EE + e0 + kh * 8;
    v8f c = {};
    c = wmma16(qa0, load_frag(kp), c);
    c = wmma16(qa1, load_frag(kp + 32), c);
#pragma unroll
    for (int r = 0; r < 8; ++r) l[r] += __expf(c[r] * SCALE - m[r]);
  }
#pragma unroll
  for (int mask = 1; mask <= 8; mask <<= 1) {
#pragma unroll
    for (int r = 0; r < 8; ++r) l[r] += __shfl_xor(l[r], mask, 32);
  }

  if (lr == 0) {
#pragma unroll
    for (int r = 0; r < 8; ++r) {
      int row = i0 + r + kh * 8;
      ml[((long long)b * HH + h) * SS + row] = make_float2(m[r], 1.f / l[r]);
    }
  }
}

// ---------- pass 2: ew_mean (f16), masked ew (f16), 1/num_neighbors ----------
__global__ __launch_bounds__(32) void attn_edges(
    const _Float16* __restrict__ Qh, const _Float16* __restrict__ Kh,
    const float2* __restrict__ ml,
    _Float16* __restrict__ ewh, _Float16* __restrict__ maskedh,
    float* __restrict__ inv_nn) {
  int it = blockIdx.x, b = blockIdx.y;
  int lane = threadIdx.x;
  int lr = lane & 15, kh = lane >> 4;
  const _Float16* Qb = Qh + (long long)b * SS * EE;
  const _Float16* Kb = Kh + (long long)b * SS * EE;
  int i0 = it * 16;
  v16h  qa[HH][2];
  float rm[HH][8], ri[HH][8];
#pragma unroll
  for (int h = 0; h < HH; ++h) {
    const _Float16* qp = Qb + (long long)(i0 + lr) * EE + h * HD + kh * 8;
    qa[h][0] = load_frag(qp);
    qa[h][1] = load_frag(qp + 32);
#pragma unroll
    for (int r = 0; r < 8; ++r) {
      int row = i0 + r + kh * 8;
      float2 v = ml[((long long)b * HH + h) * SS + row];
      rm[h][r] = v.x; ri[h][r] = v.y;
    }
  }
  float cnt[8] = {0.f, 0.f, 0.f, 0.f, 0.f, 0.f, 0.f, 0.f};
  for (int j0 = 0; j0 < SS; j0 += 16) {
    v8f ew = {};
#pragma unroll
    for (int h = 0; h < HH; ++h) {
      const _Float16* kp = Kb + (long long)(j0 + lr) * EE + h * HD + kh * 8;
      v8f c = {};
      c = wmma16(qa[h][0], load_frag(kp), c);
      c = wmma16(qa[h][1], load_frag(kp + 32), c);
#pragma unroll
      for (int r = 0; r < 8; ++r)
        ew[r] += __expf(c[r] * SCALE - rm[h][r]) * ri[h][r];
    }
#pragma unroll
    for (int r = 0; r < 8; ++r) {
      float e   = ew[r] * (1.0f / (float)HH);
      int   row = i0 + r + kh * 8;
      int   col = j0 + lr;
      float adj = (e > THRESH) ? 1.f : 0.f;
      cnt[r] += adj;
      long long idx = ((long long)b * SS + row) * SS + col;
      ewh[idx]     = (_Float16)e;
      maskedh[idx] = (_Float16)(adj != 0.f ? e : 0.f);
    }
  }
#pragma unroll
  for (int mask = 1; mask <= 8; mask <<= 1) {
#pragma unroll
    for (int r = 0; r < 8; ++r) cnt[r] += __shfl_xor(cnt[r], mask, 32);
  }
  if (lr == 0) {
#pragma unroll
    for (int r = 0; r < 8; ++r) {
      int row = i0 + r + kh * 8;
      inv_nn[(long long)b * SS + row] = 1.f / fmaxf(cnt[r], 1.f);
    }
  }
}

// ---------- generic WMMA GEMM: C = A @ W^T (+bias)(+gelu)(+add0*rowscale+add1) ----------
// W operand is always row-major [N][K] (weights, or pre-transposed activations),
// so every fragment load is two contiguous 16-byte vector loads.
template <bool GELU>
__global__ __launch_bounds__(128) void gemm_wmma(
    const _Float16* __restrict__ A, long long sAb, int lda,
    const _Float16* __restrict__ Bm, long long sBb, int ldb,
    const float* __restrict__ bias,
    const float* __restrict__ add0, const float* __restrict__ rowscale,
    const float* __restrict__ add1,
    float* __restrict__ Cf, _Float16* __restrict__ Ch, long long sCb,
    int N, int K) {
  int lane = threadIdx.x & 31;
  int wave = threadIdx.x >> 5;
  int lr = lane & 15, kh = lane >> 4;
  int b  = blockIdx.z;
  int m0 = (blockIdx.y * 4 + wave) * 16;
  int n0 = blockIdx.x * 64;
  const _Float16* Ab = A + (long long)b * sAb;
  const _Float16* Bb = Bm + (long long)b * sBb;
  v8f acc[4] = {};
  for (int k0 = 0; k0 < K; k0 += 32) {
    v16h a = load_frag(Ab + (long long)(m0 + lr) * lda + k0 + kh * 8);
#pragma unroll
    for (int t = 0; t < 4; ++t) {
      v16h bf = load_frag(Bb + (long long)(n0 + t * 16 + lr) * ldb + k0 + kh * 8);
      acc[t] = wmma16(a, bf, acc[t]);
    }
  }
#pragma unroll
  for (int t = 0; t < 4; ++t) {
    int col = n0 + t * 16 + lr;
    float bv = bias ? bias[col] : 0.f;
#pragma unroll
    for (int r = 0; r < 8; ++r) {
      int row = m0 + r + kh * 8;
      float v = acc[t][r] + bv;
      if (GELU) v = 0.5f * v * (1.f + erff(v * 0.7071067811865475f));
      long long idx = (long long)row * N + col;  // add0/add1 only used when gridDim.z==1
      if (add0) v += add0[idx] * (rowscale ? rowscale[row] : 1.f);
      if (add1) v += add1[idx];
      long long cidx = (long long)b * sCb + idx;
      if (Cf) Cf[cidx] = v;
      if (Ch) Ch[cidx] = (_Float16)v;
    }
  }
}

extern "C" void kernel_launch(void* const* d_in, const int* in_sizes, int n_in,
                              void* d_out, int out_size, void* d_ws, size_t ws_size,
                              hipStream_t stream) {
  (void)in_sizes; (void)n_in; (void)out_size; (void)ws_size;
  const float* x   = (const float*)d_in[0];
  const float* Wq  = (const float*)d_in[1];
  const float* bq  = (const float*)d_in[2];
  const float* Wk  = (const float*)d_in[3];
  const float* bk  = (const float*)d_in[4];
  const float* Wv  = (const float*)d_in[5];
  const float* bv  = (const float*)d_in[6];
  const float* Wm1 = (const float*)d_in[7];
  const float* bm1 = (const float*)d_in[8];
  const float* Wm2 = (const float*)d_in[9];
  const float* bm2 = (const float*)d_in[10];
  const float* Wo  = (const float*)d_in[11];
  const float* bo  = (const float*)d_in[12];
  float* out = (float*)d_out;

  const long long BS = (long long)BB * SS;  // 4096
  char* base = (char*)d_ws;
  size_t off = 0;
  auto carve = [&](size_t bytes) -> void* {
    void* p = base + off;
    off = (off + bytes + 255) & ~(size_t)255;
    return p;
  };
  _Float16* xh   = (_Float16*)carve((size_t)BS * EE * 2);
  _Float16* xhT  = (_Float16*)carve((size_t)BS * EE * 2);  // [B][E][S]
  _Float16* Qh   = (_Float16*)carve((size_t)BS * EE * 2);
  _Float16* Kh   = (_Float16*)carve((size_t)BS * EE * 2);
  _Float16* Vh   = (_Float16*)carve((size_t)BS * EE * 2);
  _Float16* VhT  = (_Float16*)carve((size_t)BS * EE * 2);  // [B][E][S]
  _Float16* Wqh  = (_Float16*)carve((size_t)EE * EE * 2);
  _Float16* Wkh  = (_Float16*)carve((size_t)EE * EE * 2);
  _Float16* Wvh  = (_Float16*)carve((size_t)EE * EE * 2);
  _Float16* Wm1h = (_Float16*)carve((size_t)EE * EE * 2);
  _Float16* Wm2h = (_Float16*)carve((size_t)EE * EE * 2);
  _Float16* Woh  = (_Float16*)carve((size_t)EE * EE * 2);
  float*    bm1e = (float*)carve((size_t)EE * 4);
  float2*   ml   = (float2*)carve((size_t)BB * HH * SS * sizeof(float2));
  _Float16* ewh  = (_Float16*)carve((size_t)BB * SS * SS * 2);
  _Float16* mkh  = (_Float16*)carve((size_t)BB * SS * SS * 2);
  float* inv_nn  = (float*)carve((size_t)BS * 4);
  float* AGG     = (float*)carve((size_t)BS * EE * 4);
  float* ATT     = (float*)carve((size_t)BS * EE * 4);
  _Float16* Hg   = (_Float16*)carve((size_t)BS * EE * 2);
  _Float16* Zh   = (_Float16*)carve((size_t)BS * EE * 2);

  // f32 -> f16 conversions
  {
    long long n = BS * EE;
    cvt_f16<<<(int)((n + 255) / 256), 256, 0, stream>>>(x, xh, (int)BS, EE, EE);
  }
  {
    long long n = (long long)EE * EE;
    int g = (int)((n + 255) / 256);
    cvt_f16<<<g, 256, 0, stream>>>(Wq,  Wqh,  EE, EE, EE);
    cvt_f16<<<g, 256, 0, stream>>>(Wk,  Wkh,  EE, EE, EE);
    cvt_f16<<<g, 256, 0, stream>>>(Wv,  Wvh,  EE, EE, EE);
    cvt_f16<<<g, 256, 0, stream>>>(Wm1, Wm1h, EE, EE, EE + HH);  // first 512 cols
    cvt_f16<<<g, 256, 0, stream>>>(Wm2, Wm2h, EE, EE, EE);
    cvt_f16<<<g, 256, 0, stream>>>(Wo,  Woh,  EE, EE, EE);
  }
  make_bm1eff<<<2, 256, 0, stream>>>(Wm1, bm1, bm1e);

  dim3 blk(128);
  dim3 gbig(EE / 64, (int)(BS / 64), 1);
  dim3 gtr(EE / 32, SS / 32, BB);

  // x^T per batch: [S][E] -> [E][S]
  transpose_f16<<<gtr, 256, 0, stream>>>(xh, xhT, SS, EE);

  // Q / K / V projections -> f16
  gemm_wmma<false><<<gbig, blk, 0, stream>>>(xh, 0, EE, Wqh, 0, EE, bq,
      nullptr, nullptr, nullptr, nullptr, Qh, 0, EE, EE);
  gemm_wmma<false><<<gbig, blk, 0, stream>>>(xh, 0, EE, Wkh, 0, EE, bk,
      nullptr, nullptr, nullptr, nullptr, Kh, 0, EE, EE);
  gemm_wmma<false><<<gbig, blk, 0, stream>>>(xh, 0, EE, Wvh, 0, EE, bv,
      nullptr, nullptr, nullptr, nullptr, Vh, 0, EE, EE);

  // V^T per batch
  transpose_f16<<<gtr, 256, 0, stream>>>(Vh, VhT, SS, EE);

  // attention: stats, then edge weights / adjacency
  attn_stats<<<dim3(SS / 16, HH, BB), dim3(32), 0, stream>>>(Qh, Kh, ml);
  attn_edges<<<dim3(SS / 16, BB), dim3(32), 0, stream>>>(Qh, Kh, ml, ewh, mkh, inv_nn);

  // aggregated = masked @ x ; attended = ew @ v   (batched; B = pre-transposed acts)
  dim3 gatt(EE / 64, SS / 64, BB);
  gemm_wmma<false><<<gatt, blk, 0, stream>>>(mkh, (long long)SS * SS, SS,
      xhT, (long long)EE * SS, SS, nullptr, nullptr, nullptr, nullptr,
      AGG, nullptr, (long long)SS * EE, EE, SS);
  gemm_wmma<false><<<gatt, blk, 0, stream>>>(ewh, (long long)SS * SS, SS,
      VhT, (long long)EE * SS, SS, nullptr, nullptr, nullptr, nullptr,
      ATT, nullptr, (long long)SS * EE, EE, SS);

  // MLP: Hg = gelu(x @ Wm1'^T + bm1eff)
  gemm_wmma<true><<<gbig, blk, 0, stream>>>(xh, 0, EE, Wm1h, 0, EE, bm1e,
      nullptr, nullptr, nullptr, nullptr, Hg, 0, EE, EE);
  // Z = Hg @ Wm2^T + bm2 + AGG*inv_nn[row] + ATT   -> f16
  gemm_wmma<false><<<gbig, blk, 0, stream>>>(Hg, 0, EE, Wm2h, 0, EE, bm2,
      AGG, inv_nn, ATT, nullptr, Zh, 0, EE, EE);
  // out = Z @ Wo^T + bo  -> f32
  gemm_wmma<false><<<gbig, blk, 0, stream>>>(Zh, 0, EE, Woh, 0, EE, bo,
      nullptr, nullptr, nullptr, out, nullptr, 0, EE, EE);
}
